// PANLSAGGDecoder_31404800868837
// MI455X (gfx1250) — matrix operand order repaired and checked by reading, not compiled, not run on hardware
//
#include <hip/hip_runtime.h>

// ---------------------------------------------------------------------------
// CDNA5 (gfx1250) PAN-LSAGG decoder forward — NHWC-f16 activation pipeline.
// WMMA f32<-f16 GEMMs with fully contiguous fragment loads (2xB128 for B,
// immediate K offsets), b128 NHWC stores, padded channel strides so the
// K-loop is guard-free. Deterministic two-stage BN stats. wave32 throughout.
// ---------------------------------------------------------------------------

typedef __attribute__((ext_vector_type(16))) _Float16 v16h;
typedef __attribute__((ext_vector_type(8)))  _Float16 v8h;
typedef __attribute__((ext_vector_type(8)))  float    v8f;

#define EPSV 1e-5f

#define ST_NHWC_F16 0
#define ST_NHWC_F32 1
#define ST_NCHW_F32 2

// ===========================================================================
// Pack f32 weights (M x K) into zero-padded f16 (Mcover x Kp), row-major.
// ===========================================================================
__global__ __launch_bounds__(256)
void k_pack_a(const float* __restrict__ A, _Float16* __restrict__ Ap,
              int M, int K, int Mcover, int Kp)
{
    const int i = blockIdx.x * 256 + threadIdx.x;
    if (i >= Mcover * Kp) return;
    const int m = i / Kp, k = i % Kp;
    const float v = (m < M && k < K) ? A[(long)m * K + k] : 0.f;
    Ap[i] = (_Float16)v;
}

// ===========================================================================
// GEMM over NHWC activations:  C[n, m] = sum_k A[m,k] * B[n,k]  (+bias)
// A: packed f16 (Mcover x Kp) row-major.  B: f16 NHWC, channel stride Kp.
// One wave computes two 16x16 M-tiles sharing one B fragment.
// Inner loop: 6x global_load_b128 with immediate offsets, 2x v_wmma. No
// guards, no address arithmetic (Kp*2 bytes <= 768 fits the 24-bit offset).
// Store modes: NHWC f16 (b128), NHWC f32 (2x b128), NCHW f32 (heads).
// ===========================================================================
__global__ __launch_bounds__(128)
void k_gemm_nhwc(const _Float16* __restrict__ A, const _Float16* __restrict__ B,
                 const float* __restrict__ bias, void* __restrict__ Cv,
                 int M, int Mstore, int Kp, int CpOut,
                 int mode, int HWshift, long HWmask)
{
    const int wave = threadIdx.x >> 5;
    const int lane = threadIdx.x & 31;
    const long n   = (long)blockIdx.x * 64 + wave * 16 + (lane & 15);
    const int  m0  = blockIdx.y * 32;
    const int  mr  = lane & 15;
    const int  hi  = lane >> 4;

    const _Float16* __restrict__ A0 = A + (long)(m0 + mr) * Kp + hi * 8;
    const _Float16* __restrict__ A1 = A0 + (long)16 * Kp;
    const _Float16* __restrict__ Bp = B + n * Kp + hi * 8;

    v8f acc0 = {}, acc1 = {};
    for (int k0 = 0; k0 < Kp; k0 += 32) {
        // A: halves 0..7 = K(k0+hi*8 .. +7), halves 8..15 = +16 (ISA layout)
        const v8h a0l = *(const v8h*)(A0 + k0);
        const v8h a0h = *(const v8h*)(A0 + k0 + 16);
        const v8h a1l = *(const v8h*)(A1 + k0);
        const v8h a1h = *(const v8h*)(A1 + k0 + 16);
        // B: halves 0..7 = K(k0+hi*8 .. +7), halves 8..15 = +16 (ISA layout)
        const v8h bl  = *(const v8h*)(Bp + k0);
        const v8h bh  = *(const v8h*)(Bp + k0 + 16);

        const v16h af0 = __builtin_shufflevector(a0l, a0h,
            0,1,2,3,4,5,6,7,8,9,10,11,12,13,14,15);
        const v16h af1 = __builtin_shufflevector(a1l, a1h,
            0,1,2,3,4,5,6,7,8,9,10,11,12,13,14,15);
        const v16h bf  = __builtin_shufflevector(bl, bh,
            0,1,2,3,4,5,6,7,8,9,10,11,12,13,14,15);

        acc0 = __builtin_amdgcn_wmma_f32_16x16x32_f16(
            false, af0, false, bf, (short)0, acc0, false, false);
        acc1 = __builtin_amdgcn_wmma_f32_16x16x32_f16(
            false, af1, false, bf, (short)0, acc1, false, false);
    }

    const int base0 = m0 + hi * 8;        // D: vgpr r -> M = base + r
    const int base1 = base0 + 16;

    if (mode == ST_NHWC_F16) {
        _Float16* Cb = (_Float16*)Cv + n * CpOut;
        if (base0 < Mstore) {
            v8h o;
#pragma unroll
            for (int r = 0; r < 8; ++r) {
                const int mo = base0 + r;
                float v = 0.f;
                if (mo < M) { v = acc0[r]; if (bias) v += bias[mo]; }
                o[r] = (_Float16)v;
            }
            *(v8h*)(Cb + base0) = o;
        }
        if (base1 < Mstore) {
            v8h o;
#pragma unroll
            for (int r = 0; r < 8; ++r) {
                const int mo = base1 + r;
                float v = 0.f;
                if (mo < M) { v = acc1[r]; if (bias) v += bias[mo]; }
                o[r] = (_Float16)v;
            }
            *(v8h*)(Cb + base1) = o;
        }
    } else if (mode == ST_NHWC_F32) {
        float* Cb = (float*)Cv + n * CpOut;
        if (base0 < Mstore) {
            float4 lo, hc;
            float t[8];
#pragma unroll
            for (int r = 0; r < 8; ++r) {
                const int mo = base0 + r;
                float v = 0.f;
                if (mo < M) { v = acc0[r]; if (bias) v += bias[mo]; }
                t[r] = v;
            }
            lo = make_float4(t[0], t[1], t[2], t[3]);
            hc = make_float4(t[4], t[5], t[6], t[7]);
            *(float4*)(Cb + base0)     = lo;
            *(float4*)(Cb + base0 + 4) = hc;
        }
        if (base1 < Mstore) {
            float t[8];
#pragma unroll
            for (int r = 0; r < 8; ++r) {
                const int mo = base1 + r;
                float v = 0.f;
                if (mo < M) { v = acc1[r]; if (bias) v += bias[mo]; }
                t[r] = v;
            }
            *(float4*)(Cb + base1)     = make_float4(t[0], t[1], t[2], t[3]);
            *(float4*)(Cb + base1 + 4) = make_float4(t[4], t[5], t[6], t[7]);
        }
    } else { // ST_NCHW_F32 (heads, M == Mstore == 3)
        float* Cb = (float*)Cv;
        const long b = n >> HWshift;
        const long p = n & HWmask;
#pragma unroll
        for (int r = 0; r < 8; ++r) {
            const int mo = base0 + r;
            if (mo < Mstore) {
                float v = acc0[r];
                if (bias) v += bias[mo];
                Cb[((b * Mstore + mo) << HWshift) + p] = v;
            }
        }
    }
}

// ===========================================================================
// Deterministic two-stage BN stats over NHWC (PTOT pixels, Cp stride).
// Stage 1: grid (64, Cp/32), block 256 = 32 channels x 8 pixel-lanes.
// Stage 2: grid (Creal), block 64: reduce 64 partials -> mean / inv-std.
// ===========================================================================
__global__ __launch_bounds__(256)
void k_stats1_f16(const _Float16* __restrict__ x, float* __restrict__ part,
                  int Cp, long PTOT)
{
    const int  c  = (threadIdx.x & 31) + blockIdx.y * 32;
    const int  pl = threadIdx.x >> 5;
    const long PS = PTOT >> 6;                 // gridDim.x == 64
    const long p0 = (long)blockIdx.x * PS;
    float s = 0.f, q = 0.f;
    for (long p = p0 + pl; p < p0 + PS; p += 8) {
        const float v = (float)x[p * Cp + c];
        s += v; q += v * v;
    }
    __shared__ float ss[256], sq[256];
    ss[threadIdx.x] = s; sq[threadIdx.x] = q;
    __syncthreads();
    for (int off = 128; off >= 32; off >>= 1) {
        if ((int)threadIdx.x < off) {
            ss[threadIdx.x] += ss[threadIdx.x + off];
            sq[threadIdx.x] += sq[threadIdx.x + off];
        }
        __syncthreads();
    }
    if (threadIdx.x < 32) {
        const long o = ((long)blockIdx.x * Cp + c) * 2;
        part[o] = ss[threadIdx.x]; part[o + 1] = sq[threadIdx.x];
    }
}

__global__ __launch_bounds__(256)
void k_stats1_f32(const float* __restrict__ x, float* __restrict__ part,
                  int Cp, long PTOT)
{
    const int  c  = (threadIdx.x & 31) + blockIdx.y * 32;
    const int  pl = threadIdx.x >> 5;
    const long PS = PTOT >> 6;
    const long p0 = (long)blockIdx.x * PS;
    float s = 0.f, q = 0.f;
    for (long p = p0 + pl; p < p0 + PS; p += 8) {
        const float v = x[p * Cp + c];
        s += v; q += v * v;
    }
    __shared__ float ss[256], sq[256];
    ss[threadIdx.x] = s; sq[threadIdx.x] = q;
    __syncthreads();
    for (int off = 128; off >= 32; off >>= 1) {
        if ((int)threadIdx.x < off) {
            ss[threadIdx.x] += ss[threadIdx.x + off];
            sq[threadIdx.x] += sq[threadIdx.x + off];
        }
        __syncthreads();
    }
    if (threadIdx.x < 32) {
        const long o = ((long)blockIdx.x * Cp + c) * 2;
        part[o] = ss[threadIdx.x]; part[o + 1] = sq[threadIdx.x];
    }
}

__global__ __launch_bounds__(64)
void k_stats2(const float* __restrict__ part, float* __restrict__ mean,
              float* __restrict__ istd, int Cp, long PTOT)
{
    const int c = blockIdx.x;
    __shared__ float ss[64], sq[64];
    const long o = ((long)threadIdx.x * Cp + c) * 2;
    ss[threadIdx.x] = part[o]; sq[threadIdx.x] = part[o + 1];
    __syncthreads();
    for (int off = 32; off > 0; off >>= 1) {
        if ((int)threadIdx.x < off) {
            ss[threadIdx.x] += ss[threadIdx.x + off];
            sq[threadIdx.x] += sq[threadIdx.x + off];
        }
        __syncthreads();
    }
    if (threadIdx.x == 0) {
        const float m = ss[0] / (float)PTOT;
        const float v = sq[0] / (float)PTOT - m * m;
        mean[c] = m;
        istd[c] = rsqrtf(v + EPSV);
    }
}

// BN apply (+ReLU) in place on NHWC f16; pad channels forced to 0.
// grid (PTOT/8, Cp/32), block 256.
__global__ __launch_bounds__(256)
void k_bn_apply_f16(_Float16* __restrict__ x, const float* __restrict__ mean,
                    const float* __restrict__ istd, const float* __restrict__ g,
                    const float* __restrict__ b, int Cp, int Creal)
{
    const int  c = (threadIdx.x & 31) + blockIdx.y * 32;
    const long p = (long)blockIdx.x * 8 + (threadIdx.x >> 5);
    const long i = p * Cp + c;
    float v = (float)x[i];
    if (c < Creal) v = fmaxf((v - mean[c]) * istd[c] * g[c] + b[c], 0.f);
    else           v = 0.f;
    x[i] = (_Float16)v;
}

// z = BN(o) + BN(r), f32 NHWC inputs -> f16 NHWC output.  grid (P*C/256).
__global__ __launch_bounds__(256)
void k_dualbn_f16(const float* __restrict__ o, const float* __restrict__ r,
                  _Float16* __restrict__ z,
                  const float* __restrict__ m1, const float* __restrict__ i1,
                  const float* __restrict__ g1, const float* __restrict__ b1,
                  const float* __restrict__ m2, const float* __restrict__ i2,
                  const float* __restrict__ g2, const float* __restrict__ b2,
                  int CMask)
{
    const long idx = blockIdx.x * 256L + threadIdx.x;
    const int  c   = (int)(idx & CMask);
    const float a = (o[idx] - m1[c]) * i1[c] * g1[c] + b1[c];
    const float s = (r[idx] - m2[c]) * i2[c] * g2[c] + b2[c];
    z[idx] = (_Float16)(a + s);
}

// ===========================================================================
// Depthwise 7x7 (pad 3) on NHWC f16.  grid (W/8, H, B), block = Cp threads.
// Weights held in registers; pad channels (zero weights on zero data) stay 0.
// ===========================================================================
__global__ void k_dw7_nhwc(const _Float16* __restrict__ x,
                           const float* __restrict__ w,
                           _Float16* __restrict__ y,
                           int H, int W, int C, int Cp)
{
    const int c = threadIdx.x;
    const int b = blockIdx.z, h = blockIdx.y, w0 = blockIdx.x * 8;
    float wr[49];
    const bool real = (c < C);
#pragma unroll
    for (int t = 0; t < 49; ++t) wr[t] = real ? w[c * 49 + t] : 0.f;
    const long pixBase = (long)b * H * W;
    for (int px = 0; px < 8; ++px) {
        const int ww = w0 + px;
        float acc = 0.f;
#pragma unroll
        for (int dy = 0; dy < 7; ++dy) {
            const int hy = h + dy - 3;
            if (hy < 0 || hy >= H) continue;
#pragma unroll
            for (int dx = 0; dx < 7; ++dx) {
                const int wx = ww + dx - 3;
                if (wx < 0 || wx >= W) continue;
                acc += wr[dy * 7 + dx] *
                       (float)x[(pixBase + (long)hy * W + wx) * Cp + c];
            }
        }
        y[(pixBase + (long)h * W + ww) * Cp + c] = (_Float16)acc;
    }
}

// ===========================================================================
// GroupNorm (G=8, 1 ch/group) on wg: f32 NHWC, stride 16.  grid (B), 256 thr
// = 8 channels x 32 pixel-lanes; stats + apply in one kernel.
// ===========================================================================
__global__ __launch_bounds__(256)
void k_gn_nhwc(float* __restrict__ x, const float* __restrict__ g,
               const float* __restrict__ bb, long HW)
{
    const int  b  = blockIdx.x;
    const int  c  = threadIdx.x & 7;
    const int  pl = threadIdx.x >> 3;     // 0..31
    float* base = x + (long)b * HW * 16;
    float s = 0.f, q = 0.f;
    for (long p = pl; p < HW; p += 32) {
        const float v = base[p * 16 + c];
        s += v; q += v * v;
    }
    __shared__ float ss[256], sq[256];
    __shared__ float sm[8], si[8];
    ss[threadIdx.x] = s; sq[threadIdx.x] = q;
    __syncthreads();
    for (int off = 128; off >= 8; off >>= 1) {
        if ((int)threadIdx.x < off) {
            ss[threadIdx.x] += ss[threadIdx.x + off];
            sq[threadIdx.x] += sq[threadIdx.x + off];
        }
        __syncthreads();
    }
    if (threadIdx.x < 8) {
        const float m = ss[threadIdx.x] / (float)HW;
        const float v = sq[threadIdx.x] / (float)HW - m * m;
        sm[threadIdx.x] = m; si[threadIdx.x] = rsqrtf(v + EPSV);
    }
    __syncthreads();
    const float mm = sm[c], is = si[c], gg = g[c], bt = bb[c];
    for (long p = pl; p < HW; p += 32)
        base[p * 16 + c] = (base[p * 16 + c] - mm) * is * gg + bt;
}

// ===========================================================================
// SKA gate + adaptive channel pooling (NHWC).  grid (P*Cout/256).
// ===========================================================================
__global__ __launch_bounds__(256)
void k_gate_agg_nhwc(const _Float16* __restrict__ x, const float* __restrict__ wg,
                     float* __restrict__ o, int Cin, int Cout,
                     int CoMask, int CoShift)
{
    const long idx = blockIdx.x * 256L + threadIdx.x;
    const long p   = idx >> CoShift;
    const int  oc  = (int)(idx & CoMask);
    const int  s   = (oc * Cin) / Cout;
    const int  e   = ((oc + 1) * Cin + Cout - 1) / Cout;
    const float coef = 1.f / (float)(e - s);
    const int  Cg = Cin / 8;
    const _Float16* xp = x + p * (long)Cin;
    const float*    wp = wg + p * 16;
    float acc = 0.f;
    for (int c = s; c < e; ++c) acc += (float)xp[c] * wp[c / Cg];
    o[idx] = acc * coef;
}

// ===========================================================================
// Ingest: NCHW f32 -> NHWC f16 via 32x32 LDS transpose tiles.
// grid (HW/32, C/32, B), block 256 (32x8).
// ===========================================================================
__global__ __launch_bounds__(256)
void k_nchw_to_nhwc(const float* __restrict__ src, _Float16* __restrict__ dst,
                    int C, long HW, int Cdst, int coff)
{
    __shared__ float t[32][33];
    const int  b  = blockIdx.z;
    const long p0 = (long)blockIdx.x * 32;
    const int  c0 = blockIdx.y * 32;
    const int  tx = threadIdx.x & 31, ty = threadIdx.x >> 5;
#pragma unroll
    for (int i = 0; i < 4; ++i) {
        const int cc = c0 + ty + i * 8;
        t[ty + i * 8][tx] = src[((long)b * C + cc) * HW + p0 + tx];
    }
    __syncthreads();
#pragma unroll
    for (int i = 0; i < 4; ++i) {
        const long p = p0 + ty + i * 8;
        dst[((long)b * HW + p) * Cdst + coff + c0 + tx] =
            (_Float16)t[tx][ty + i * 8];
    }
}

// NHWC f16 channel-slice copy into concat buffer.  grid (P*Csrc/256).
__global__ __launch_bounds__(256)
void k_copy_ch_nhwc(const _Float16* __restrict__ src, _Float16* __restrict__ dst,
                    int Cdst, int coff, int CsMask, int CsShift)
{
    const long idx = blockIdx.x * 256L + threadIdx.x;
    const long p   = idx >> CsShift;
    const int  c   = (int)(idx & CsMask);
    dst[p * Cdst + coff + c] = src[idx];
}

// Bilinear x2 upsample, NHWC f16.  grid (Wout, Hout, B), block = C threads.
__global__ void k_up2_nhwc(const _Float16* __restrict__ src,
                           _Float16* __restrict__ dst,
                           int C, int Hin, int Win, int Cdst, int coff)
{
    const int ox = blockIdx.x, oy = blockIdx.y, b = blockIdx.z;
    const int Wout = 2 * Win;
    const float fy = oy * 0.5f - 0.25f;
    const float fx = ox * 0.5f - 0.25f;
    const int y0 = (int)floorf(fy), x0 = (int)floorf(fx);
    const float wy = fy - (float)y0, wx = fx - (float)x0;
    const int y0c = max(y0, 0), y1c = min(y0 + 1, Hin - 1);
    const int x0c = max(x0, 0), x1c = min(x0 + 1, Win - 1);
    const int c = threadIdx.x;
    const long sb = (long)b * Hin * Win;
    const float v00 = (float)src[(sb + (long)y0c * Win + x0c) * C + c];
    const float v01 = (float)src[(sb + (long)y0c * Win + x1c) * C + c];
    const float v10 = (float)src[(sb + (long)y1c * Win + x0c) * C + c];
    const float v11 = (float)src[(sb + (long)y1c * Win + x1c) * C + c];
    const float v = (1.f - wy) * ((1.f - wx) * v00 + wx * v01) +
                    wy * ((1.f - wx) * v10 + wx * v11);
    const long dp = (long)b * (4L * Hin * Win) + (long)oy * Wout + ox;
    dst[dp * Cdst + coff + c] = (_Float16)v;
}

// 2x2 average pool, NHWC f16.  grid (Wout, Hout, B), block = C threads.
__global__ void k_down2_nhwc(const _Float16* __restrict__ src,
                             _Float16* __restrict__ dst,
                             int C, int Hout, int Wout, int Cdst, int coff)
{
    const int ox = blockIdx.x, oy = blockIdx.y, b = blockIdx.z;
    const int Win = 2 * Wout;
    const int c = threadIdx.x;
    const long sb = (long)b * (4L * Hout * Wout);
    const long p00 = sb + (long)(2 * oy) * Win + 2 * ox;
    const float v = 0.25f * ((float)src[p00 * C + c] +
                             (float)src[(p00 + 1) * C + c] +
                             (float)src[(p00 + Win) * C + c] +
                             (float)src[(p00 + Win + 1) * C + c]);
    const long dp = (long)b * Hout * Wout + (long)oy * Wout + ox;
    dst[dp * Cdst + coff + c] = (_Float16)v;
}

// ===========================================================================
// Host orchestration
// ===========================================================================
namespace {

constexpr int BATCH = 4;

struct LsgP {
    const float *cv1_w, *cv1_g, *cv1_b;
    const float *cv2_w, *cv2_g, *cv2_b;
    const float *cv3_w, *cv3_g, *cv3_b;
    const float *cv4_w, *cv4_b;
    const float *gn_g, *gn_b;
    const float *bn_g, *bn_b;
    const float *res_w, *resbn_g, *resbn_b;
};

LsgP get_lsg(void* const* d_in, int base) {
    LsgP p;
    p.cv1_w = (const float*)d_in[base + 0];
    p.cv1_g = (const float*)d_in[base + 1];
    p.cv1_b = (const float*)d_in[base + 2];
    p.cv2_w = (const float*)d_in[base + 3];
    p.cv2_g = (const float*)d_in[base + 4];
    p.cv2_b = (const float*)d_in[base + 5];
    p.cv3_w = (const float*)d_in[base + 6];
    p.cv3_g = (const float*)d_in[base + 7];
    p.cv3_b = (const float*)d_in[base + 8];
    p.cv4_w = (const float*)d_in[base + 9];
    p.cv4_b = (const float*)d_in[base + 10];
    p.gn_g  = (const float*)d_in[base + 11];
    p.gn_b  = (const float*)d_in[base + 12];
    p.bn_g  = (const float*)d_in[base + 13];
    p.bn_b  = (const float*)d_in[base + 14];
    p.res_w = (const float*)d_in[base + 15];
    p.resbn_g = (const float*)d_in[base + 16];
    p.resbn_b = (const float*)d_in[base + 17];
    return p;
}

struct Ws {
    _Float16 *xcat, *ha, *hb, *z1, *z2, *z3, *d2, *d3, *t3;
    float *wg, *oraw, *rraw;
    float *apack, *part;
    float *m1, *i1, *m2, *i2;
};

inline void run_gemm(hipStream_t s, const Ws& w, const float* A,
                     const _Float16* B, const float* bias, void* C,
                     int M, int K, int Kp, int Mcover, int Mstore, int CpOut,
                     long P, int mode, int HWshift)
{
    _Float16* Ap = (_Float16*)w.apack;
    const int tot = Mcover * Kp;
    k_pack_a<<<(tot + 255) / 256, 256, 0, s>>>(A, Ap, M, K, Mcover, Kp);
    dim3 g((unsigned)(P / 64), Mcover / 32);
    k_gemm_nhwc<<<g, 128, 0, s>>>(Ap, B, bias, C, M, Mstore, Kp, CpOut,
                                  mode, HWshift, (1L << HWshift) - 1);
}

inline void run_bn_relu_f16(hipStream_t s, const Ws& w, _Float16* x,
                            const float* g, const float* b,
                            int Creal, int Cp, long P)
{
    k_stats1_f16<<<dim3(64, Cp / 32), 256, 0, s>>>(x, w.part, Cp, P);
    k_stats2<<<Creal, 64, 0, s>>>(w.part, w.m1, w.i1, Cp, P);
    k_bn_apply_f16<<<dim3((unsigned)(P / 8), Cp / 32), 256, 0, s>>>(
        x, w.m1, w.i1, g, b, Cp, Creal);
}

static int ilog2(int v) { int s = 0; while ((1 << s) < v) ++s; return s; }

void run_lsg(hipStream_t s, const Ws& w, const LsgP& p,
             int cin, int cout, int H, int W, _Float16* zout)
{
    const int  mid  = cin / 2;
    const int  midP = (mid + 31) & ~31;
    const long HW   = (long)H * W;
    const long P    = (long)BATCH * HW;

    // cv1 1x1 -> BN -> ReLU (into ha, stride midP; pad rows zeroed)
    run_gemm(s, w, p.cv1_w, w.xcat, nullptr, w.ha,
             mid, cin, cin, midP, midP, midP, P, ST_NHWC_F16, 0);
    run_bn_relu_f16(s, w, w.ha, p.cv1_g, p.cv1_b, mid, midP, P);

    // cv2 depthwise 7x7 -> BN -> ReLU
    k_dw7_nhwc<<<dim3(W / 8, H, BATCH), midP, 0, s>>>(
        w.ha, p.cv2_w, w.hb, H, W, mid, midP);
    run_bn_relu_f16(s, w, w.hb, p.cv2_g, p.cv2_b, mid, midP, P);

    // cv3 1x1 (Kp = midP, pad rows of hb are exact zeros) -> BN -> ReLU
    run_gemm(s, w, p.cv3_w, w.hb, nullptr, w.ha,
             mid, mid, midP, midP, midP, midP, P, ST_NHWC_F16, 0);
    run_bn_relu_f16(s, w, w.ha, p.cv3_g, p.cv3_b, mid, midP, P);

    // cv4 1x1 (+bias) -> wg f32 (stride 16) -> GroupNorm
    run_gemm(s, w, p.cv4_w, w.ha, p.cv4_b, w.wg,
             8, mid, midP, 32, 16, 16, P, ST_NHWC_F32, 0);
    k_gn_nhwc<<<BATCH, 256, 0, s>>>(w.wg, p.gn_g, p.gn_b, HW);

    // gate + channel pooling -> o_raw f32 ; residual 1x1 -> r_raw f32
    const int coShift = ilog2(cout);
    k_gate_agg_nhwc<<<(unsigned)((P * cout) / 256), 256, 0, s>>>(
        w.xcat, w.wg, w.oraw, cin, cout, cout - 1, coShift);
    run_gemm(s, w, p.res_w, w.xcat, nullptr, w.rraw,
             cout, cin, cin, cout, cout, cout, P, ST_NHWC_F32, 0);

    // BN both branches (deterministic 2-stage), fused residual add -> z f16
    k_stats1_f32<<<dim3(64, cout / 32), 256, 0, s>>>(w.oraw, w.part, cout, P);
    k_stats2<<<cout, 64, 0, s>>>(w.part, w.m1, w.i1, cout, P);
    k_stats1_f32<<<dim3(64, cout / 32), 256, 0, s>>>(w.rraw, w.part, cout, P);
    k_stats2<<<cout, 64, 0, s>>>(w.part, w.m2, w.i2, cout, P);
    k_dualbn_f16<<<(unsigned)((P * cout) / 256), 256, 0, s>>>(
        w.oraw, w.rraw, zout, w.m1, w.i1, p.bn_g, p.bn_b,
        w.m2, w.i2, p.resbn_g, p.resbn_b, cout - 1);
}

} // namespace

extern "C" void kernel_launch(void* const* d_in, const int* in_sizes, int n_in,
                              void* d_out, int out_size, void* d_ws, size_t ws_size,
                              hipStream_t stream) {
    (void)in_sizes; (void)n_in; (void)out_size; (void)ws_size;

    const float* obs0 = (const float*)d_in[0]; // (4, 32,256,256)
    const float* obs1 = (const float*)d_in[1]; // (4, 64,128,128)
    const float* obs2 = (const float*)d_in[2]; // (4,128, 64, 64)
    const float* obs3 = (const float*)d_in[3]; // (4,256, 32, 32)

    const LsgP ud3 = get_lsg(d_in, 4);
    const LsgP ud2 = get_lsg(d_in, 22);
    const LsgP ud1 = get_lsg(d_in, 40);
    const LsgP dd1 = get_lsg(d_in, 58);
    const LsgP dd2 = get_lsg(d_in, 76);
    const float* h1_w = (const float*)d_in[94];
    const float* h1_b = (const float*)d_in[95];
    const float* h2_w = (const float*)d_in[96];
    const float* h2_b = (const float*)d_in[97];
    const float* h3_w = (const float*)d_in[98];
    const float* h3_b = (const float*)d_in[99];

    // ---- static workspace layout (float units) ----
    float* base = (float*)d_ws;
    Ws w;
    w.xcat  = (_Float16*)(base);               // 96ch * 262144px f16 = 12,582,912 fl
    w.ha    = (_Float16*)(base + 12582912);    // 64 * 262144 f16    =  8,388,608 fl
    w.hb    = (_Float16*)(base + 20971520);    //                       8,388,608 fl
    w.wg    = base + 29360128;                 // 16 * 262144 f32    =  4,194,304 fl
    w.oraw  = base + 33554432;                 // 32 * 262144 f32    =  8,388,608 fl
    w.rraw  = base + 41943040;                 //                       8,388,608 fl
    w.z1    = (_Float16*)(base + 50331648);    // 32 * 262144 f16    =  4,194,304 fl
    w.z2    = (_Float16*)(base + 54525952);    // 64 *  65536 f16    =  2,097,152 fl
    w.z3    = (_Float16*)(base + 56623104);    // 128 * 16384 f16    =  1,048,576 fl
    w.d2    = (_Float16*)(base + 57671680);    //                       2,097,152 fl
    w.d3    = (_Float16*)(base + 59768832);    //                       1,048,576 fl
    w.t3    = (_Float16*)(base + 60817408);    // 256 *  4096 f16    =    524,288 fl
    w.apack = base + 61341696;                 //                          40,960 fl
    w.part  = base + 61382656;                 //                          32,768 fl
    w.m1    = base + 61415424;
    w.i1    = w.m1 + 512;
    w.m2    = w.i1 + 512;
    w.i2    = w.m2 + 512;

    hipStream_t s = stream;

    // ---- ud3: z3 = LSG(concat[obs2, up2(obs3)]), cin=384, cout=128, 64x64 ----
    k_nchw_to_nhwc<<<dim3(4096 / 32, 128 / 32, BATCH), 256, 0, s>>>(
        obs2, w.xcat, 128, 4096, 384, 0);
    k_nchw_to_nhwc<<<dim3(1024 / 32, 256 / 32, BATCH), 256, 0, s>>>(
        obs3, w.t3, 256, 1024, 256, 0);
    k_up2_nhwc<<<dim3(64, 64, BATCH), 256, 0, s>>>(w.t3, w.xcat, 256, 32, 32, 384, 128);
    run_lsg(s, w, ud3, 384, 128, 64, 64, w.z3);

    // ---- ud2: z2 = LSG(concat[obs1, up2(z3)]), cin=192, cout=64, 128x128 ----
    k_nchw_to_nhwc<<<dim3(16384 / 32, 64 / 32, BATCH), 256, 0, s>>>(
        obs1, w.xcat, 64, 16384, 192, 0);
    k_up2_nhwc<<<dim3(128, 128, BATCH), 128, 0, s>>>(w.z3, w.xcat, 128, 64, 64, 192, 64);
    run_lsg(s, w, ud2, 192, 64, 128, 128, w.z2);

    // ---- ud1: z1 = LSG(concat[obs0, up2(z2)]), cin=96, cout=32, 256x256 ----
    k_nchw_to_nhwc<<<dim3(65536 / 32, 32 / 32, BATCH), 256, 0, s>>>(
        obs0, w.xcat, 32, 65536, 96, 0);
    k_up2_nhwc<<<dim3(256, 256, BATCH), 64, 0, s>>>(w.z2, w.xcat, 64, 128, 128, 96, 32);
    run_lsg(s, w, ud1, 96, 32, 256, 256, w.z1);

    // ---- heads: p1 | p2 | p3 in d_out (fp32 NCHW) ----
    float* out = (float*)d_out;
    float* p1 = out;                       // (4,3,256,256)
    float* p2 = p1 + 4L * 3 * 65536;       // (4,3,128,128)
    float* p3 = p2 + 4L * 3 * 16384;       // (4,3, 64, 64)

    run_gemm(s, w, h1_w, w.z1, h1_b, p1, 3, 32, 32, 32, 3, 0,
             4L * 65536, ST_NCHW_F32, 16);

    // ---- dd1: d2 = LSG(concat[down2(z1), z2]), cin=96, cout=64, 128x128 ----
    k_down2_nhwc<<<dim3(128, 128, BATCH), 32, 0, s>>>(w.z1, w.xcat, 32, 128, 128, 96, 0);
    k_copy_ch_nhwc<<<(unsigned)((4L * 16384 * 64) / 256), 256, 0, s>>>(
        w.z2, w.xcat, 96, 32, 63, 6);
    run_lsg(s, w, dd1, 96, 64, 128, 128, w.d2);

    run_gemm(s, w, h2_w, w.d2, h2_b, p2, 3, 64, 64, 32, 3, 0,
             4L * 16384, ST_NCHW_F32, 14);

    // ---- dd2: d3 = LSG(concat[down2(d2), z3]), cin=192, cout=128, 64x64 ----
    k_down2_nhwc<<<dim3(64, 64, BATCH), 64, 0, s>>>(w.d2, w.xcat, 64, 64, 64, 192, 0);
    k_copy_ch_nhwc<<<(unsigned)((4L * 4096 * 128) / 256), 256, 0, s>>>(
        w.z3, w.xcat, 192, 64, 127, 7);
    run_lsg(s, w, dd2, 192, 128, 64, 64, w.d3);

    run_gemm(s, w, h3_w, w.d3, h3_b, p3, 3, 128, 128, 32, 3, 0,
             4L * 4096, ST_NCHW_F32, 12);
}
